// MOEModel_88416196756290
// MI455X (gfx1250) — matrix operand order repaired and checked by reading, not compiled, not run on hardware
//
#include <hip/hip_runtime.h>
#include <hip/hip_bf16.h>
#include <stdint.h>

// Problem constants (from reference): B,N,HL,D,HID,O,E,S,K = 8,64,32,256,1024,256,8,3,2
#define Dk    256
#define HIDk  1024
#define Ok    256
#define Ek    8
#define Sk    3
#define NXP   11          // S shared + E independent expert "slots"
#define TM    64          // tokens per block in the main kernel
#define TOKS  16384       // B*N*HL

typedef __attribute__((ext_vector_type(16))) __bf16 v16bf;
typedef __attribute__((ext_vector_type(8)))  float  v8f;

union Frag32 { v16bf bf; uint4 u[2]; };   // 32 bytes: one 16x32 bf16 A/B fragment per lane

__device__ __forceinline__ unsigned short f2bf(float f) {
  union { float f; unsigned u; } cv; cv.f = f;
  unsigned u = cv.u;
  unsigned r = u + 0x7FFFu + ((u >> 16) & 1u);   // round-to-nearest-even
  return (unsigned short)(r >> 16);
}

// ---------------------------------------------------------------------------
// 1) Gating: softmax(x @ gate_w + gate_b) -> top-2 -> renorm -> dense w_full
//    One wave32 per token; 8 waves (tokens) per 256-thread block.
// ---------------------------------------------------------------------------
__global__ void __launch_bounds__(256) gate_kernel(
    const float* __restrict__ x, const float* __restrict__ gw,
    const float* __restrict__ gb, float* __restrict__ wfull) {
  const int lane = threadIdx.x & 31;
  const int t = blockIdx.x * 8 + (threadIdx.x >> 5);

  float p[Ek];
#pragma unroll
  for (int e = 0; e < Ek; ++e) p[e] = 0.f;

#pragma unroll
  for (int j = 0; j < Dk / 32; ++j) {
    const int d = lane + 32 * j;
    const float xv = x[(size_t)t * Dk + d];
#pragma unroll
    for (int e = 0; e < Ek; ++e) p[e] += xv * gw[d * Ek + e];
  }
  // wave32 cross-lane reduction
#pragma unroll
  for (int e = 0; e < Ek; ++e) {
#pragma unroll
    for (int off = 16; off > 0; off >>= 1) p[e] += __shfl_xor(p[e], off, 32);
    p[e] += gb[e];
  }
  // softmax numerators (denominator cancels in top-2 renormalization)
  float mx = p[0];
#pragma unroll
  for (int e = 1; e < Ek; ++e) mx = fmaxf(mx, p[e]);
#pragma unroll
  for (int e = 0; e < Ek; ++e) p[e] = __expf(p[e] - mx);

  int i0 = 0; float v0 = p[0];
#pragma unroll
  for (int e = 1; e < Ek; ++e) if (p[e] > v0) { v0 = p[e]; i0 = e; }
  int i1 = -1; float v1 = -1.f;
#pragma unroll
  for (int e = 0; e < Ek; ++e) if (e != i0 && p[e] > v1) { v1 = p[e]; i1 = e; }
  const float inv = 1.f / (v0 + v1);

  if (lane < Ek) {
    float w = (lane == i0) ? v0 * inv : ((lane == i1) ? v1 * inv : 0.f);
    wfull[(size_t)t * Ek + lane] = w;
  }
}

// ---------------------------------------------------------------------------
// 2) fp32 -> bf16 conversion (x) and transpose+convert (weights)
// ---------------------------------------------------------------------------
__global__ void __launch_bounds__(256) cvt_bf16_kernel(
    const float* __restrict__ src, unsigned short* __restrict__ dst, int n) {
  for (int i = blockIdx.x * blockDim.x + threadIdx.x; i < n;
       i += gridDim.x * blockDim.x)
    dst[i] = f2bf(src[i]);
}

// src: (nExp, R, C) f32 row-major  ->  dst: (nExp, C, R) bf16  (dst[e][c][r] = src[e][r][c])
__global__ void __launch_bounds__(256) transpose_cvt_kernel(
    const float* __restrict__ src, unsigned short* __restrict__ dst,
    int nExp, int R, int C) {
  const int total = nExp * R * C;
  for (int i = blockIdx.x * blockDim.x + threadIdx.x; i < total;
       i += gridDim.x * blockDim.x) {
    const int e = i / (R * C);
    const int rc = i - e * (R * C);
    const int r = rc / C, c = rc - r * C;
    dst[(size_t)e * R * C + (size_t)c * R + r] = f2bf(src[i]);
  }
}

// ---------------------------------------------------------------------------
// 3) Main fused MoE kernel. One block = 64 tokens (4 M-subtiles of 16).
//    For each of 11 expert slots:  h = relu(x @ W1 + b1)  ->  o = h @ W2 + b2,
//    accumulate per-row weighted o into persistent f32 registers; store 0.5*acc.
//
//    Fragment layouts used by this kernel (wave32, 16-bit; A and B share one k order):
//      A (16x32, MxK): lane = M | hi<<4; 16 contiguous K per lane:
//                                        u[0..1] = K[hi*16 .. hi*16+15]
//      B (32x16, KxN): lane = N | hi<<4; 16 CONTIGUOUS K per lane:
//                                        u[0..1] = K[hi*16 .. hi*16+15]
//      C/D (16x16 f32): VGPR r = row (r + 8*hi), lane%16 = column.
// ---------------------------------------------------------------------------
__global__ void __launch_bounds__(256) moe_main_kernel(
    const unsigned short* __restrict__ xb,    // T x D bf16
    const unsigned short* __restrict__ w1T,   // NXP x HID x D bf16 (transposed)
    const unsigned short* __restrict__ w2T,   // NXP x O  x HID bf16 (transposed)
    const float* __restrict__ sh_b1, const float* __restrict__ sh_b2,
    const float* __restrict__ in_b1, const float* __restrict__ in_b2,
    const float* __restrict__ obj,            // T x S
    const float* __restrict__ wfull,          // T x E
    float* __restrict__ out) {                // T x O
  extern __shared__ char smem[];
  unsigned short* sX = (unsigned short*)smem;                        // TM x D   bf16 (32KB)
  unsigned short* sH = (unsigned short*)(smem + TM * Dk * 2);        // TM x HID bf16 (128KB)
  float* sW = (float*)(smem + TM * Dk * 2 + TM * HIDk * 2);          // NXP x TM f32

  const int tile = blockIdx.x;
  const int t0 = tile * TM;
  const int tid = threadIdx.x;
  const int lane = tid & 31;
  const int wv = tid >> 5;          // 8 waves
  const int col = lane & 15;        // fragment M (for A) or N (for B/C) selector
  const int hi = lane >> 4;         // K-half selector per documented layouts

  // stage x tile into LDS (contiguous 32KB block)
  {
    const uint4* src = (const uint4*)(xb + (size_t)t0 * Dk);
    uint4* dst = (uint4*)sX;
    const int nvec = TM * Dk / 8;
    for (int i = tid; i < nvec; i += 256) dst[i] = src[i];
  }
  // stage per-token combine weights: slots 0..2 = obj, 3..10 = gate weights
  for (int i = tid; i < NXP * TM; i += 256) {
    const int xp = i / TM, m = i - xp * TM;
    const int t = t0 + m;
    sW[i] = (xp < Sk) ? obj[(size_t)t * Sk + xp]
                      : wfull[(size_t)t * Ek + (xp - Sk)];
  }
  __syncthreads();

  v8f acc[2][4];
  const v8f vz = {};
#pragma unroll
  for (int ot = 0; ot < 2; ++ot)
#pragma unroll
    for (int ms = 0; ms < 4; ++ms) acc[ot][ms] = vz;

  for (int xp = 0; xp < NXP; ++xp) {
    const unsigned short* w1 = w1T + (size_t)xp * HIDk * Dk;   // (HID, D)
    const unsigned short* w2 = w2T + (size_t)xp * Ok * HIDk;   // (O, HID)
    const float* b1 = (xp < Sk) ? (sh_b1 + xp * HIDk) : (in_b1 + (xp - Sk) * HIDk);
    const float* b2 = (xp < Sk) ? (sh_b2 + xp * Ok)   : (in_b2 + (xp - Sk) * Ok);

    // ---- GEMM1: (64 x 256) @ (256 x 1024), waves split the 64 HID N-tiles ----
    for (int nt = 0; nt < 8; ++nt) {
      const int nrow = (wv * 8 + nt) * 16 + col;   // HID column owned by this lane
      v8f c[4];
#pragma unroll
      for (int ms = 0; ms < 4; ++ms) c[ms] = vz;

      for (int ks = 0; ks < Dk / 32; ++ks) {
        const int k = ks * 32;
        // B fragment: 16 contiguous K values per lane (32-byte region)
        Frag32 bf;
        const unsigned short* br = w1 + (size_t)nrow * Dk + k + hi * 16;
        bf.u[0] = *(const uint4*)br;
        bf.u[1] = *(const uint4*)(br + 8);
#pragma unroll
        for (int ms = 0; ms < 4; ++ms) {
          // A fragment: 16 contiguous K values per lane {hi*16..+15}
          Frag32 af;
          const unsigned short* ar = sX + (size_t)(ms * 16 + col) * Dk + k + hi * 16;
          af.u[0] = *(const uint4*)ar;
          af.u[1] = *(const uint4*)(ar + 8);
          c[ms] = __builtin_amdgcn_wmma_f32_16x16x32_bf16(
              false, af.bf, false, bf.bf, (short)0, c[ms], false, false);
        }
      }
      const float bias = b1[nrow];
#pragma unroll
      for (int ms = 0; ms < 4; ++ms) {
#pragma unroll
        for (int r = 0; r < 8; ++r) {
          float v = c[ms][r] + bias;
          v = v > 0.f ? v : 0.f;                       // relu
          sH[(size_t)(ms * 16 + r + hi * 8) * HIDk + nrow] = f2bf(v);
        }
      }
    }
    __syncthreads();

    // ---- GEMM2: (64 x 1024) @ (1024 x 256), waves split the 16 O N-tiles ----
#pragma unroll
    for (int ot = 0; ot < 2; ++ot) {
      const int orow = (wv * 2 + ot) * 16 + col;     // O column owned by this lane
      v8f c[4];
#pragma unroll
      for (int ms = 0; ms < 4; ++ms) c[ms] = vz;

      for (int ks = 0; ks < HIDk / 32; ++ks) {
        const int k = ks * 32;
        Frag32 bf;
        const unsigned short* br = w2 + (size_t)orow * HIDk + k + hi * 16;
        bf.u[0] = *(const uint4*)br;
        bf.u[1] = *(const uint4*)(br + 8);
#pragma unroll
        for (int ms = 0; ms < 4; ++ms) {
          Frag32 af;
          const unsigned short* ar = sH + (size_t)(ms * 16 + col) * HIDk + k + hi * 16;
          af.u[0] = *(const uint4*)ar;
          af.u[1] = *(const uint4*)(ar + 8);
          c[ms] = __builtin_amdgcn_wmma_f32_16x16x32_bf16(
              false, af.bf, false, bf.bf, (short)0, c[ms], false, false);
        }
      }
      const float bias2 = b2[orow];
#pragma unroll
      for (int ms = 0; ms < 4; ++ms) {
#pragma unroll
        for (int r = 0; r < 8; ++r) {
          const float sc = sW[xp * TM + ms * 16 + r + hi * 8];
          acc[ot][ms][r] += sc * (c[ms][r] + bias2);
        }
      }
    }
    __syncthreads();   // sH reused next slot
  }

  // final 0.5*(shared + independent) and store
#pragma unroll
  for (int ot = 0; ot < 2; ++ot) {
    const int orow = (wv * 2 + ot) * 16 + col;
#pragma unroll
    for (int ms = 0; ms < 4; ++ms) {
#pragma unroll
      for (int r = 0; r < 8; ++r) {
        const int t = t0 + ms * 16 + r + hi * 8;
        out[(size_t)t * Ok + orow] = 0.5f * acc[ot][ms][r];
      }
    }
  }
}

// ---------------------------------------------------------------------------
extern "C" void kernel_launch(void* const* d_in, const int* in_sizes, int n_in,
                              void* d_out, int out_size, void* d_ws, size_t ws_size,
                              hipStream_t stream) {
  const float* x   = (const float*)d_in[0];
  const float* obj = (const float*)d_in[1];
  const float* gw  = (const float*)d_in[2];
  const float* gb  = (const float*)d_in[3];
  const float* sw1 = (const float*)d_in[4];
  const float* sb1 = (const float*)d_in[5];
  const float* sw2 = (const float*)d_in[6];
  const float* sb2 = (const float*)d_in[7];
  const float* iw1 = (const float*)d_in[8];
  const float* ib1 = (const float*)d_in[9];
  const float* iw2 = (const float*)d_in[10];
  const float* ib2 = (const float*)d_in[11];
  float* out = (float*)d_out;

  const int T = in_sizes[0] / Dk;     // 16384

  // workspace layout (bytes)
  char* ws = (char*)d_ws;
  unsigned short* xb   = (unsigned short*)ws;                              // T*D bf16
  size_t off = (size_t)T * Dk * 2;
  unsigned short* w1T  = (unsigned short*)(ws + off);                      // NXP*HID*D bf16
  off += (size_t)NXP * HIDk * Dk * 2;
  unsigned short* w2T  = (unsigned short*)(ws + off);                      // NXP*O*HID bf16
  off += (size_t)NXP * Ok * HIDk * 2;
  float* wfull = (float*)(ws + off);                                      // T*E f32
  off += (size_t)T * Ek * 4;
  (void)ws_size; (void)n_in; (void)out_size;

  // 1) gating
  gate_kernel<<<T / 8, 256, 0, stream>>>(x, gw, gb, wfull);

  // 2) conversions / transposes
  cvt_bf16_kernel<<<2048, 256, 0, stream>>>(x, xb, T * Dk);
  transpose_cvt_kernel<<<1024, 256, 0, stream>>>(sw1, w1T,                 Sk, Dk, HIDk);
  transpose_cvt_kernel<<<2048, 256, 0, stream>>>(iw1, w1T + (size_t)Sk * HIDk * Dk, Ek, Dk, HIDk);
  transpose_cvt_kernel<<<1024, 256, 0, stream>>>(sw2, w2T,                 Sk, HIDk, Ok);
  transpose_cvt_kernel<<<2048, 256, 0, stream>>>(iw2, w2T + (size_t)Sk * Ok * HIDk, Ek, HIDk, Ok);

  // 3) fused MoE (~163KB dynamic LDS; CDNA5 WGP allows up to 320KB per workgroup)
  const size_t lds = (size_t)TM * Dk * 2 + (size_t)TM * HIDk * 2 + (size_t)NXP * TM * 4;
  (void)hipFuncSetAttribute((const void*)moe_main_kernel,
                            hipFuncAttributeMaxDynamicSharedMemorySize, (int)lds);
  moe_main_kernel<<<T / TM, 256, lds, stream>>>(
      xb, w1T, w2T, sb1, sb2, ib1, ib2, obj, wfull, out);
}
